// CAttention_12034498363841
// MI455X (gfx1250) — matrix operand order repaired and checked by reading, not compiled, not run on hardware
//
#include <hip/hip_runtime.h>
#include <hip/hip_bf16.h>
#include <stdint.h>
#include <math.h>

typedef __attribute__((ext_vector_type(16))) __bf16 v16bf;
typedef __attribute__((ext_vector_type(8)))  float  v8f;

#define HEADS    8
#define DIM_HEAD 64
#define MAXP     512
#define BB       4
#define NN       1024
#define DIMM     512
#define ROWS     (BB * NN)      // 4096
#define KCOMB    1024           // [Xre ; Xim] K for projections
#define NQKV     3072           // qr | qi | (kr vr) | (ki vi)
#define RLD      1088           // 1025 rel columns padded to 17*64
#define SCALE    0.125f         // 64^-0.5
#define BTLD     36             // padded k-stride of transposed B tile (bank-conflict break)

// ---------------- CDNA5 async LDS-DMA helpers (ASYNCcnt path, ISA §10.7/§15.18) ----------------

__device__ __forceinline__ void async_load_b128(uint32_t lds_addr, uint32_t gofs, const void* sbase) {
    // GVS mode: mem_addr = SGPR64 + VGPR32 + inst_offset ; copies 16B straight into LDS.
    asm volatile("global_load_async_to_lds_b128 %0, %1, %2"
                 :: "v"(lds_addr), "v"(gofs), "s"(sbase)
                 : "memory");
}

__device__ __forceinline__ void wait_async() {
    asm volatile("s_wait_asynccnt 0x0" ::: "memory");
}

// ---------------- WMMA fragment helpers (CDNA5 §7.12.2 layouts) ----------------

__device__ __forceinline__ v8f wmma_bf16(v16bf a, v16bf b, v8f c) {
    return __builtin_amdgcn_wmma_f32_16x16x32_bf16(
        /*neg_a=*/false, a, /*neg_b=*/false, b,
        /*c_mod=*/(short)0, c, /*reuse_a=*/false, /*reuse_b=*/false);
}

// A fragment: 16x32 bf16, row m = lane&15, K pattern per ISA table.
__device__ __forceinline__ v16bf load_frag_a(const __bf16* p, int ld) {
    const int lane = threadIdx.x & 31;
    const int m = lane & 15;
    const int h = lane >> 4;
    v16bf f;
#pragma unroll
    for (int e = 0; e < 16; ++e) {
        const int v = e >> 1, q = e & 1;
        const int k = (v < 4 ? 2 * v : 16 + 2 * (v - 4)) + 8 * h + q;
        f[e] = p[m * ld + k];
    }
    return f;
}

__device__ __forceinline__ v16bf load_frag_a_g(const __bf16* p, long ld) {
    const int lane = threadIdx.x & 31;
    const int m = lane & 15;
    const int h = lane >> 4;
    v16bf f;
#pragma unroll
    for (int e = 0; e < 16; ++e) {
        const int v = e >> 1, q = e & 1;
        const int k = (v < 4 ? 2 * v : 16 + 2 * (v - 4)) + 8 * h + q;
        f[e] = p[(long)m * ld + k];
    }
    return f;
}

// B fragment: 32x16 bf16. Element e -> K per ISA table (e<8: e + 8*(lane>>4); else 16 + 8*(lane>>4) + e-8), N = lane&15.
// Element (k,n) at p[k*sk + n*sn]; with sk==1 each lane reads two runs of 16 contiguous
// bytes -> vectorizes to ds_load_b128.
__device__ __forceinline__ v16bf load_frag_b(const __bf16* p, int sk, int sn) {
    const int lane = threadIdx.x & 31;
    const int n = lane & 15;
    const int h = lane >> 4;
    v16bf f;
#pragma unroll
    for (int e = 0; e < 16; ++e) {
        const int k = (e < 8 ? e : 8 + e) + 8 * h;
        f[e] = p[k * sk + n * sn];
    }
    return f;
}

// ---------------- pack kernels (fp32 -> bf16, complex-combined blocks) ----------------

__global__ void pack_A_x(const float* __restrict__ x, __bf16* __restrict__ A) {
    int idx = blockIdx.x * blockDim.x + threadIdx.x;      // 4096*1024
    int row = idx >> 10, k = idx & 1023;
    float v = (k < DIMM) ? x[((long)row * DIMM + k) * 2]
                         : x[((long)row * DIMM + (k - DIMM)) * 2 + 1];
    A[idx] = (__bf16)v;
}

__global__ void pack_B_qkv(const float* __restrict__ wq_r, const float* __restrict__ wq_i,
                           const float* __restrict__ wkv_r, const float* __restrict__ wkv_i,
                           __bf16* __restrict__ Bm) {
    long idx = (long)blockIdx.x * blockDim.x + threadIdx.x;   // 1024*3072
    int k = (int)(idx / NQKV), n = (int)(idx % NQKV);
    int lo = (k < DIMM);
    int kk = lo ? k : k - DIMM;
    float v;
    if (n < 512)        v = lo ?  wq_r[(long)kk * 512 + n]          : -wq_i[(long)kk * 512 + n];
    else if (n < 1024)  v = lo ?  wq_i[(long)kk * 512 + (n - 512)]  :  wq_r[(long)kk * 512 + (n - 512)];
    else if (n < 2048)  v = lo ?  wkv_r[(long)kk * 1024 + (n - 1024)] : -wkv_i[(long)kk * 1024 + (n - 1024)];
    else                v = lo ?  wkv_i[(long)kk * 1024 + (n - 2048)] :  wkv_r[(long)kk * 1024 + (n - 2048)];
    Bm[idx] = (__bf16)v;
}

__global__ void pack_Brel(const float* __restrict__ rel, __bf16* __restrict__ Bm) {
    int idx = blockIdx.x * blockDim.x + threadIdx.x;      // 64*1088
    int d = idx / RLD, p = idx % RLD;
    float v = (p < 2 * MAXP + 1) ? rel[(long)p * DIM_HEAD + d] : 0.f;
    Bm[idx] = (__bf16)v;
}

__global__ void pack_B_fin(const float* __restrict__ wo_r, const float* __restrict__ wo_i,
                           __bf16* __restrict__ Bm) {
    long idx = (long)blockIdx.x * blockDim.x + threadIdx.x;   // 1024*1024
    int k = (int)(idx >> 10), n = (int)(idx & 1023);
    int lo = (k < DIMM);
    int kk = lo ? k : k - DIMM;
    float v;
    if (n < 512) v = lo ? wo_r[(long)kk * 512 + n]         : -wo_i[(long)kk * 512 + n];
    else         v = lo ? wo_i[(long)kk * 512 + (n - 512)] :  wo_r[(long)kk * 512 + (n - 512)];
    Bm[idx] = (__bf16)v;
}

__global__ void pack_bias(const float* __restrict__ bo_r, const float* __restrict__ bo_i,
                          float* __restrict__ bias) {
    int n = blockIdx.x * blockDim.x + threadIdx.x;        // 1024
    bias[n] = (n < DIMM) ? bo_r[n] : bo_i[n - DIMM];
}

// ---------------- generic tiled WMMA GEMM: C[M,N] = A[M,K] * B[K,N] ----------------
// 128 threads = 4 waves; block tile 64x64; wave w owns rows [16w,16w+16).
// A tile is DMA'd into LDS via global_load_async_to_lds_b128; B tile is staged
// TRANSPOSED (sBt[n][k], padded stride) so B-fragment gathers are contiguous.
// MODE 0: store bf16 C.  MODE 2: fp32 + bias, interleaved [row][col%512][col/512].

template <int MODE>
__global__ __launch_bounds__(128) void gemm_wmma(
    const __bf16* __restrict__ A, int lda, long strideA,
    const __bf16* __restrict__ Bm, int ldb, long strideB,
    void* __restrict__ Cout, int ldc, long strideC,
    int Kdim, const float* __restrict__ bias)
{
    __shared__ __align__(16) __bf16   sA[64 * 32];       // row-major 64 x 32
    __shared__ __align__(16) uint16_t sBt[64 * BTLD];    // transposed: n rows x k cols (padded)
    const int t = threadIdx.x, wave = t >> 5, lane = t & 31;
    const int n0 = blockIdx.x * 64, m0 = blockIdx.y * 64;
    const int z = blockIdx.z;
    A  += (long)z * strideA;
    Bm += (long)z * strideB;

    v8f acc[4] = {};
    const uint32_t* gb = (const uint32_t*)Bm;
    const uint32_t ldsA = (uint32_t)(uintptr_t)sA;

    for (int k0 = 0; k0 < Kdim; k0 += 32) {
        // A tile 64x32 bf16 = 256 x 16B chunks: async LDS-DMA, 2 per thread.
#pragma unroll
        for (int i = 0; i < 2; ++i) {
            int idx = t + i * 128;                       // 0..255
            int r = idx >> 2, c = idx & 3;               // row, 16B chunk in row
            uint32_t gofs = (uint32_t)(((long)(m0 + r) * lda + k0) * 2) + c * 16;
            async_load_b128(ldsA + idx * 16, gofs, A);
        }
        // B tile 32x64 bf16, stored transposed with padded stride.
#pragma unroll
        for (int i = 0; i < 8; ++i) {
            int idx = t + i * 128;                       // 0..1023 u32 chunks
            int rb = idx >> 5, cb = idx & 31;            // k row, u32 col (2 n's)
            uint32_t w = gb[((long)(k0 + rb) * ldb + n0) / 2 + cb];
            sBt[(2 * cb)     * BTLD + rb] = (uint16_t)(w & 0xffffu);
            sBt[(2 * cb + 1) * BTLD + rb] = (uint16_t)(w >> 16);
        }
        wait_async();
        __syncthreads();
        v16bf a = load_frag_a(sA + wave * 16 * 32, 32);
#pragma unroll
        for (int nt = 0; nt < 4; ++nt) {
            v16bf b = load_frag_b((const __bf16*)(sBt + nt * 16 * BTLD), /*sk=*/1, /*sn=*/BTLD);
            acc[nt] = wmma_bf16(a, b, acc[nt]);
        }
        __syncthreads();
    }

    const int half = lane >> 4, nlo = lane & 15;
#pragma unroll
    for (int nt = 0; nt < 4; ++nt) {
#pragma unroll
        for (int v = 0; v < 8; ++v) {
            int row = m0 + wave * 16 + v + 8 * half;
            int col = n0 + nt * 16 + nlo;
            float val = acc[nt][v];
            if (MODE == 0) {
                __bf16* C = (__bf16*)Cout;
                C[(long)z * strideC + (long)row * ldc + col] = (__bf16)val;
            } else {
                float* C = (float*)Cout;
                val += bias[col];
                C[((long)row * DIMM + (col & 511)) * 2 + (col >> 9)] = val;
            }
        }
    }
}

// ---------------- fused complex attention (flash-style, magnitude softmax) ----------------
// grid = (1024/128, HEADS, B), block = 256 (8 waves). Wave owns 16 query rows.
// K/V j-blocks are DMA'd to LDS with global_load_async_to_lds_b128 (ASYNCcnt).

__global__ __launch_bounds__(256) void attn_kernel(
    const __bf16* __restrict__ qkv,   // [4096][3072]
    const __bf16* __restrict__ Rr,    // [8][4096][RLD]
    const __bf16* __restrict__ Ri,
    __bf16* __restrict__ afin)        // [4096][1024]  (out_r | out_i)
{
    __shared__ __align__(16) __bf16 sKr[64 * 64], sKi[64 * 64], sVr[64 * 64], sVi[64 * 64];
    __shared__ __align__(16) __bf16 sP[8][16 * 64];

    const int t = threadIdx.x, wave = t >> 5, lane = t & 31;
    const int half = lane >> 4, nlo = lane & 15;
    const int hd = blockIdx.y;
    const int bb = blockIdx.z;
    const int i0 = blockIdx.x * 128 + wave * 16;
    const long rowbase = (long)bb * NN;

    const int qoff_r = hd * 64, qoff_i = 512 + hd * 64;
    const int koff_r = 1024 + hd * 64, voff_r = 1536 + hd * 64;
    const int koff_i = 2048 + hd * 64, voff_i = 2560 + hd * 64;

    v16bf Qr[2], Qi[2];
    {
        const __bf16* qbase = qkv + (rowbase + i0) * NQKV;
        Qr[0] = load_frag_a_g(qbase + qoff_r,      NQKV);
        Qr[1] = load_frag_a_g(qbase + qoff_r + 32, NQKV);
        Qi[0] = load_frag_a_g(qbase + qoff_i,      NQKV);
        Qi[1] = load_frag_a_g(qbase + qoff_i + 32, NQKV);
    }

    v8f Or[4] = {}, Oi[4] = {};
    float mrow[8], lrow[8];
#pragma unroll
    for (int v = 0; v < 8; ++v) { mrow[v] = -1e30f; lrow[v] = 0.f; }

    const __bf16* Rrp = Rr + ((long)hd * ROWS + rowbase) * RLD;
    const __bf16* Rip = Ri + ((long)hd * ROWS + rowbase) * RLD;

    const uint32_t ldsKV[4] = {(uint32_t)(uintptr_t)sKr, (uint32_t)(uintptr_t)sKi,
                               (uint32_t)(uintptr_t)sVr, (uint32_t)(uintptr_t)sVi};
    const int offs[4] = {koff_r, koff_i, voff_r, voff_i};

    for (int j0 = 0; j0 < NN; j0 += 64) {
        // ---- async K/V block DMA: 4 mats x 8KB = 512 x 16B chunks each ----
#pragma unroll
        for (int mm = 0; mm < 4; ++mm) {
#pragma unroll
            for (int i = 0; i < 2; ++i) {
                int idx = t + i * 256;                   // 0..511
                int r = idx >> 3, c = idx & 7;           // row, 16B chunk (8 per 128B row)
                uint32_t gofs = (uint32_t)((rowbase + j0 + r) * (NQKV * 2))
                              + (uint32_t)(offs[mm] * 2) + c * 16;
                async_load_b128(ldsKV[mm] + idx * 16, gofs, qkv);
            }
        }
        wait_async();
        __syncthreads();

        // ---- complex scores: Sr = QrKr^T - QiKi^T, Si = QrKi^T + QiKr^T ----
        float sc[4][8];
#pragma unroll
        for (int jt = 0; jt < 4; ++jt) {
            const __bf16* kb = sKr + jt * 16 * 64;
            const __bf16* ib = sKi + jt * 16 * 64;
            v16bf bKr0 = load_frag_b(kb,      1, 64);    // K^T gather from row-major K tile
            v16bf bKr1 = load_frag_b(kb + 32, 1, 64);
            v16bf bKi0 = load_frag_b(ib,      1, 64);
            v16bf bKi1 = load_frag_b(ib + 32, 1, 64);
            v8f srA = {}, srB = {}, siA = {}, siB = {};
            srA = wmma_bf16(Qr[0], bKr0, srA); srA = wmma_bf16(Qr[1], bKr1, srA);
            srB = wmma_bf16(Qi[0], bKi0, srB); srB = wmma_bf16(Qi[1], bKi1, srB);
            siA = wmma_bf16(Qr[0], bKi0, siA); siA = wmma_bf16(Qr[1], bKi1, siA);
            siB = wmma_bf16(Qi[0], bKr0, siB); siB = wmma_bf16(Qi[1], bKr1, siB);
#pragma unroll
            for (int v = 0; v < 8; ++v) {
                int irow = i0 + v + 8 * half;
                int jcol = j0 + jt * 16 + nlo;
                int p = irow - jcol;
                p = p > MAXP ? MAXP : (p < -MAXP ? -MAXP : p);
                p += MAXP;
                float ra = (float)Rrp[(long)irow * RLD + p];
                float ia = (float)Rip[(long)irow * RLD + p];
                float sr = (srA[v] - srB[v] + ra) * SCALE;
                float si = (siA[v] + siB[v] + ia) * SCALE;
                sc[jt][v] = sqrtf(sr * sr + si * si);
            }
        }

        // ---- online softmax over this 64-key block ----
        float alpha[8];
#pragma unroll
        for (int v = 0; v < 8; ++v) {
            float m = sc[0][v];
#pragma unroll
            for (int jt = 1; jt < 4; ++jt) m = fmaxf(m, sc[jt][v]);
#pragma unroll
            for (int off = 1; off < 16; off <<= 1) m = fmaxf(m, __shfl_xor(m, off, 32));
            float mnew = fmaxf(mrow[v], m);
            alpha[v] = __expf(mrow[v] - mnew);
            mrow[v] = mnew;
            float rs = 0.f;
#pragma unroll
            for (int jt = 0; jt < 4; ++jt) {
                float e = __expf(sc[jt][v] - mnew);
                sc[jt][v] = e;
                rs += e;
            }
#pragma unroll
            for (int off = 1; off < 16; off <<= 1) rs += __shfl_xor(rs, off, 32);
            lrow[v] = lrow[v] * alpha[v] + rs;
        }
#pragma unroll
        for (int dt = 0; dt < 4; ++dt)
#pragma unroll
            for (int v = 0; v < 8; ++v) { Or[dt][v] *= alpha[v]; Oi[dt][v] *= alpha[v]; }

        // ---- P (C-layout) -> LDS -> A-fragments ----
        __bf16* Pw = sP[wave];
#pragma unroll
        for (int jt = 0; jt < 4; ++jt)
#pragma unroll
            for (int v = 0; v < 8; ++v)
                Pw[(v + 8 * half) * 64 + jt * 16 + nlo] = (__bf16)sc[jt][v];
        __syncthreads();

        v16bf Pa0 = load_frag_a(Pw, 64);
        v16bf Pa1 = load_frag_a(Pw + 32, 64);
#pragma unroll
        for (int dt = 0; dt < 4; ++dt) {
            v16bf bVr0 = load_frag_b(sVr + dt * 16,           64, 1);
            v16bf bVr1 = load_frag_b(sVr + 32 * 64 + dt * 16, 64, 1);
            v16bf bVi0 = load_frag_b(sVi + dt * 16,           64, 1);
            v16bf bVi1 = load_frag_b(sVi + 32 * 64 + dt * 16, 64, 1);
            Or[dt] = wmma_bf16(Pa0, bVr0, Or[dt]); Or[dt] = wmma_bf16(Pa1, bVr1, Or[dt]);
            Oi[dt] = wmma_bf16(Pa0, bVi0, Oi[dt]); Oi[dt] = wmma_bf16(Pa1, bVi1, Oi[dt]);
        }
        __syncthreads();
    }

    // ---- normalize and store into A_fin = [out_r | out_i] bf16 ----
#pragma unroll
    for (int dt = 0; dt < 4; ++dt) {
#pragma unroll
        for (int v = 0; v < 8; ++v) {
            long row = rowbase + i0 + v + 8 * half;
            int d = dt * 16 + nlo;
            float inv = 1.f / lrow[v];
            afin[row * 1024 + hd * 64 + d]       = (__bf16)(Or[dt][v] * inv);
            afin[row * 1024 + 512 + hd * 64 + d] = (__bf16)(Oi[dt][v] * inv);
        }
    }
}

// ---------------- host-side orchestration ----------------

extern "C" void kernel_launch(void* const* d_in, const int* in_sizes, int n_in,
                              void* d_out, int out_size, void* d_ws, size_t ws_size,
                              hipStream_t stream) {
    const float* x      = (const float*)d_in[0];
    const float* wq_r   = (const float*)d_in[1];
    const float* wq_i   = (const float*)d_in[2];
    const float* wkv_r  = (const float*)d_in[3];
    const float* wkv_i  = (const float*)d_in[4];
    const float* wo_r   = (const float*)d_in[5];
    const float* wo_i   = (const float*)d_in[6];
    const float* bo_r   = (const float*)d_in[7];
    const float* bo_i   = (const float*)d_in[8];
    const float* rel    = (const float*)d_in[9];
    float* out = (float*)d_out;

    char* ws = (char*)d_ws;
    size_t off = 0;
    auto alloc = [&](size_t bytes) { char* p = ws + off; off += (bytes + 255) & ~(size_t)255; return p; };
    __bf16* A_x   = (__bf16*)alloc((size_t)ROWS * KCOMB * 2);         // 8 MB
    __bf16* B_qkv = (__bf16*)alloc((size_t)KCOMB * NQKV * 2);         // 6 MB
    __bf16* QKV   = (__bf16*)alloc((size_t)ROWS * NQKV * 2);          // 24 MB
    __bf16* Brel  = (__bf16*)alloc((size_t)DIM_HEAD * RLD * 2);       // 136 KB
    __bf16* Rr    = (__bf16*)alloc((size_t)HEADS * ROWS * RLD * 2);   // 68 MB
    __bf16* Ri    = (__bf16*)alloc((size_t)HEADS * ROWS * RLD * 2);   // 68 MB
    __bf16* A_fin = (__bf16*)alloc((size_t)ROWS * 1024 * 2);          // 8 MB
    __bf16* B_fin = (__bf16*)alloc((size_t)1024 * 1024 * 2);          // 2 MB
    float*  biasC = (float*)alloc(1024 * 4);
    (void)ws_size; (void)in_sizes; (void)n_in; (void)out_size;

    // 1. pack bf16 operands
    pack_A_x  <<<(ROWS * KCOMB) / 256, 256, 0, stream>>>(x, A_x);
    pack_B_qkv<<<(KCOMB * NQKV) / 256, 256, 0, stream>>>(wq_r, wq_i, wkv_r, wkv_i, B_qkv);
    pack_Brel <<<(DIM_HEAD * RLD) / 256, 256, 0, stream>>>(rel, Brel);
    pack_B_fin<<<(1024 * 1024) / 256, 256, 0, stream>>>(wo_r, wo_i, B_fin);
    pack_bias <<<4, 256, 0, stream>>>(bo_r, bo_i, biasC);

    // 2. QKV projection: [4096,1024] x [1024,3072] -> bf16 [4096,3072]
    gemm_wmma<0><<<dim3(NQKV / 64, ROWS / 64, 1), 128, 0, stream>>>(
        A_x, KCOMB, 0, B_qkv, NQKV, 0, QKV, NQKV, 0, KCOMB, nullptr);

    // 3. relative-position GEMMs, batched over heads (z): R = Q_h x rel_emb^T
    gemm_wmma<0><<<dim3(RLD / 64, ROWS / 64, HEADS), 128, 0, stream>>>(
        QKV, NQKV, /*strideA=*/64, Brel, RLD, 0,
        Rr, RLD, /*strideC=*/(long)ROWS * RLD, DIM_HEAD, nullptr);
    gemm_wmma<0><<<dim3(RLD / 64, ROWS / 64, HEADS), 128, 0, stream>>>(
        QKV + 512, NQKV, /*strideA=*/64, Brel, RLD, 0,
        Ri, RLD, /*strideC=*/(long)ROWS * RLD, DIM_HEAD, nullptr);

    // 4. fused complex flash attention with magnitude softmax
    attn_kernel<<<dim3(NN / 128, HEADS, BB), 256, 0, stream>>>(QKV, Rr, Ri, A_fin);

    // 5. output projection + bias + interleave into [b,n,512,2] fp32
    gemm_wmma<2><<<dim3(1024 / 64, ROWS / 64, 1), 128, 0, stream>>>(
        A_fin, 1024, 0, B_fin, 1024, 0, out, 0, 0, 1024, biasC);
}
